// NodeRNN_28381143892383
// MI455X (gfx1250) — compile-verified, matrix operand order checked by reading, not run on hardware
//
#include <hip/hip_runtime.h>
#include <hip/hip_bf16.h>

typedef __bf16 bf16;
typedef __attribute__((ext_vector_type(16))) __bf16 v16bf;
typedef __attribute__((ext_vector_type(8)))  float  v8f;

#define EH   256   // EDGE_H
#define NH   128   // NODE_H
#define EMB  64
#define LDSK 272   // padded row stride (bf16 elems): multiple of 16 -> 32B aligned rows

__device__ __forceinline__ float sigmoidf_(float x) {
    return 1.0f / (1.0f + __expf(-x));
}

// One-time weight prep: combined [512 x 256] bf16 = [W_ih | W_hh] rows,
// W_hid -> bf16 [64 x 512], combined gate bias b_ih + b_hh (f32).
__global__ __launch_bounds__(256) void prep_weights(
    const float* __restrict__ W_ih, const float* __restrict__ W_hh,
    const float* __restrict__ b_ih, const float* __restrict__ b_hh,
    const float* __restrict__ W_hid,
    bf16* __restrict__ Wc, bf16* __restrict__ Whid_b, float* __restrict__ biasC)
{
    int t = blockIdx.x * 256 + threadIdx.x;     // 0 .. 131071
    int g = t >> 8;                             // gate row 0..511
    int k = t & 255;                            // K 0..255
    float w = (k < NH) ? W_ih[g * NH + k] : W_hh[g * NH + (k - NH)];
    Wc[g * 256 + k] = (bf16)w;
    if (t < EMB * 512) Whid_b[t] = (bf16)W_hid[t];
    if (t < 512)       biasC[t] = b_ih[t] + b_hh[t];
}

__global__ __launch_bounds__(256) void node_rnn_step(
    const float* __restrict__ Hv,  const float* __restrict__ hvv,
    const float* __restrict__ xv,  const float* __restrict__ hvp,
    const float* __restrict__ cvp, const int*   __restrict__ tmask,
    const float* __restrict__ Wpos, const float* __restrict__ bpos,
    const float* __restrict__ bhid,
    const bf16* __restrict__ Wc, const bf16* __restrict__ Whid,
    const float* __restrict__ biasC,
    float* __restrict__ outH, float* __restrict__ outC)
{
    __shared__ __bf16 Alds[16 * LDSK];          // A panel: [e_v | a_v | hv] bf16, 16 rows x 256 (padded)

    const int tid   = threadIdx.x;
    const int wave  = tid >> 5;
    const int lane  = tid & 31;
    const int lrow  = lane & 15;                // M (or N) index within a 16-tile
    const int khalf = (lane >> 4) << 4;         // 0 or 16: K-half owned by this lane
    const long rowBase = (long)blockIdx.x * 16;

    if (wave >= 4) {
        // ---- waves 4-7: e_v (K=2 matmul) into cols [0,64), hv_tm1 bf16 into cols [128,256)
        const int tt = tid - 128;               // 0..127
#pragma unroll
        for (int j = 0; j < 8; ++j) {
            int idx = tt + 128 * j;             // 0..1023
            int r = idx >> 6, c = idx & 63;
            long gr = rowBase + r;
            float x0 = xv[gr * 2 + 0], x1 = xv[gr * 2 + 1];
            float e = fmaf(x0, Wpos[2 * c], fmaf(x1, Wpos[2 * c + 1], bpos[c]));
            Alds[r * LDSK + c] = (bf16)fmaxf(e, 0.0f);
        }
#pragma unroll
        for (int j = 0; j < 16; ++j) {
            int idx = tt + 128 * j;             // 0..2047
            int r = idx >> 7, c = idx & 127;
            Alds[r * LDSK + 128 + c] = (bf16)hvp[(rowBase + r) * NH + c];
        }
    } else {
        // ---- waves 0-3: GEMM1 a_v tile (16 rows x 16 cols), K = 512 over [hvv | Hv]
        v8f acc = {};
        const long gr = rowBase + lrow;
#pragma unroll 4
        for (int kk = 0; kk < 16; ++kk) {
            int kb = kk * 32 + khalf;           // this lane's 16-K chunk
            const float* ap = (kb < EH) ? (hvv + gr * EH + kb)
                                        : (Hv  + gr * EH + (kb - EH));
            const float4* ap4 = (const float4*)ap;
            float4 q0 = ap4[0], q1 = ap4[1], q2 = ap4[2], q3 = ap4[3];
            v16bf a;
            a[0]=(bf16)q0.x;  a[1]=(bf16)q0.y;  a[2]=(bf16)q0.z;  a[3]=(bf16)q0.w;
            a[4]=(bf16)q1.x;  a[5]=(bf16)q1.y;  a[6]=(bf16)q1.z;  a[7]=(bf16)q1.w;
            a[8]=(bf16)q2.x;  a[9]=(bf16)q2.y;  a[10]=(bf16)q2.z; a[11]=(bf16)q2.w;
            a[12]=(bf16)q3.x; a[13]=(bf16)q3.y; a[14]=(bf16)q3.z; a[15]=(bf16)q3.w;
            v16bf b = *(const v16bf*)(Whid + (wave * 16 + lrow) * 512 + kb);
            acc = __builtin_amdgcn_wmma_f32_16x16x32_bf16(
                      false, a, false, b, (short)0, acc, false, false);
        }
        // bias + relu, write bf16 a_v into cols [64,128)
#pragma unroll
        for (int r = 0; r < 8; ++r) {
            int m = (lane < 16) ? r : r + 8;
            int c = wave * 16 + lrow;
            float v = acc[r] + bhid[c];
            Alds[m * LDSK + 64 + c] = (bf16)fmaxf(v, 0.0f);
        }
    }
    __syncthreads();

    // ---- GEMM2: gates (16 x 512), K=256 from LDS A panel.
    // Wave w owns gate columns {16w+128*G + 0..15 : G=0..3} -> i/f/g/o for hidden slice [16w,16w+16)
    v8f acc0 = {}, acc1 = {}, acc2 = {}, acc3 = {};
    const int h = wave * 16 + lrow;             // hidden index 0..127 owned by this lane's column
#pragma unroll 2
    for (int kk = 0; kk < 8; ++kk) {
        int kb = kk * 32 + khalf;
        v16bf a  = *(const v16bf*)&Alds[lrow * LDSK + kb];
        v16bf b0 = *(const v16bf*)(Wc + (size_t)(h          ) * 256 + kb);
        v16bf b1 = *(const v16bf*)(Wc + (size_t)(h + 1 * NH ) * 256 + kb);
        v16bf b2 = *(const v16bf*)(Wc + (size_t)(h + 2 * NH ) * 256 + kb);
        v16bf b3 = *(const v16bf*)(Wc + (size_t)(h + 3 * NH ) * 256 + kb);
        acc0 = __builtin_amdgcn_wmma_f32_16x16x32_bf16(false, a, false, b0, (short)0, acc0, false, false);
        acc1 = __builtin_amdgcn_wmma_f32_16x16x32_bf16(false, a, false, b1, (short)0, acc1, false, false);
        acc2 = __builtin_amdgcn_wmma_f32_16x16x32_bf16(false, a, false, b2, (short)0, acc2, false, false);
        acc3 = __builtin_amdgcn_wmma_f32_16x16x32_bf16(false, a, false, b3, (short)0, acc3, false, false);
    }

    // ---- LSTM pointwise, entirely in registers per wave
    const float bi = biasC[h], bf_ = biasC[NH + h], bg = biasC[2 * NH + h], bo = biasC[3 * NH + h];
#pragma unroll
    for (int r = 0; r < 8; ++r) {
        int m = (lane < 16) ? r : r + 8;
        long gr = rowBase + m;
        float iG = sigmoidf_(acc0[r] + bi);
        float fG = sigmoidf_(acc1[r] + bf_);
        float gG = tanhf    (acc2[r] + bg);
        float oG = sigmoidf_(acc3[r] + bo);
        float co = cvp[gr * NH + h];
        float cn = fmaf(fG, co, iG * gG);
        float hn = oG * tanhf(cn);
        int   mk = tmask[gr];
        float ho = hvp[gr * NH + h];
        outH[gr * NH + h] = mk ? hn : ho;
        outC[gr * NH + h] = mk ? cn : co;
    }
}

extern "C" void kernel_launch(void* const* d_in, const int* in_sizes, int n_in,
                              void* d_out, int out_size, void* d_ws, size_t ws_size,
                              hipStream_t stream) {
    const float* Hv    = (const float*)d_in[0];
    const float* hvv   = (const float*)d_in[1];
    const float* xvt   = (const float*)d_in[2];
    const float* hvp   = (const float*)d_in[3];
    const float* cvp   = (const float*)d_in[4];
    const int*   tmask = (const int*)  d_in[5];
    const float* W_pos = (const float*)d_in[6];
    const float* b_pos = (const float*)d_in[7];
    const float* W_hid = (const float*)d_in[8];
    const float* b_hid = (const float*)d_in[9];
    const float* W_ih  = (const float*)d_in[10];
    const float* b_ih  = (const float*)d_in[11];
    const float* W_hh  = (const float*)d_in[12];
    const float* b_hh  = (const float*)d_in[13];

    const int N = in_sizes[0] / EH;             // 262144

    char* ws = (char*)d_ws;
    bf16*  Wc     = (bf16*) (ws);               // 512*256*2 = 262144 B
    bf16*  Whid_b = (bf16*) (ws + 262144);      //  64*512*2 =  65536 B
    float* biasC  = (float*)(ws + 327680);      //  512*4    =   2048 B

    float* outH = (float*)d_out;
    float* outC = outH + (size_t)N * NH;

    prep_weights<<<512, 256, 0, stream>>>(W_ih, W_hh, b_ih, b_hh, W_hid, Wc, Whid_b, biasC);
    node_rnn_step<<<N / 16, 256, 0, stream>>>(Hv, hvv, xvt, hvp, cvp, tmask,
                                              W_pos, b_pos, b_hid, Wc, Whid_b, biasC,
                                              outH, outC);
}
